// CustomAtt_RNNAgent_53197464928656
// MI455X (gfx1250) — matrix operand order repaired and checked by reading, not compiled, not run on hardware
//
#include <hip/hip_runtime.h>

typedef __attribute__((ext_vector_type(16))) _Float16 v16h;
typedef __attribute__((ext_vector_type(8)))  float    v8f;
typedef __attribute__((ext_vector_type(4)))  unsigned u32x4;
typedef __attribute__((ext_vector_type(8)))  int      i32x8;
typedef __attribute__((ext_vector_type(4)))  int      i32x4;
typedef _Float16 h16;

#define NHEAD  4
#define DHEAD  16
#define F_OWN  40
#define FFEAT  24
#define NAL    15
#define NEN    16
#define BNTOT  32768
#define TB     16

// ---------------- address-space helpers ----------------------------------
typedef const __attribute__((address_space(1))) v16h*  gv16p;
typedef const __attribute__((address_space(1))) float* gfp;
typedef const __attribute__((address_space(3))) void*  lvp;

__device__ __forceinline__ float gload(const float* p) { return *(gfp)p; }
__device__ __forceinline__ unsigned lds_addr_of(const void* p) {
  return (unsigned)(size_t)(lvp)p;
}

// ---------------- fragment-ordered weight storage in d_ws (half units) ----
constexpr int OFF_HYPW1 = 0;                        // 4 nt * 2 kf
constexpr int OFF_HYPW2 = OFF_HYPW1 + 4*2*512;      // 164 nt * 2 kf
constexpr int OFF_WA    = OFF_HYPW2 + 164*2*512;    // 4 nt * 1 kf
constexpr int OFF_WE    = OFF_WA    + 4*1*512;
constexpr int OFF_AWQ   = OFF_WE    + 4*1*512;      // 4 nt * 2 kf each
constexpr int OFF_AWK   = OFF_AWQ   + 4*2*512;
constexpr int OFF_AWV   = OFF_AWK   + 4*2*512;
constexpr int OFF_AWO   = OFF_AWV   + 4*2*512;
constexpr int OFF_EWQ   = OFF_AWO   + 4*2*512;
constexpr int OFF_EWK   = OFF_EWQ   + 4*2*512;
constexpr int OFF_EWV   = OFF_EWK   + 4*2*512;
constexpr int OFF_EWO   = OFF_EWV   + 4*2*512;
constexpr int OFF_GWIH  = OFF_EWO   + 4*2*512;      // 12 nt * 6 kf
constexpr int OFF_GWHH  = OFF_GWIH  + 12*6*512;     // 12 nt * 2 kf
constexpr int OFF_WM    = OFF_GWHH  + 12*2*512;     // 1 nt * 2 kf
constexpr int OFF_WZK   = OFF_WM    + 1*2*512;
constexpr int OFF_WEKW  = OFF_WZK   + 4*2*512;

// ---------------- LDS layout (float units), 15616 floats = 61KB ----------
constexpr int L_OWN  = 0;     // 16 x 64 (cols >=40 zero)
constexpr int L_SCR  = 1024;  // 16 x 64 staging / A-frag source
constexpr int L_OWNE = 2048;  // 16 x 64 own_e
constexpr int L_Q    = 3072;  // 16 x 64 q / zk
constexpr int L_U    = 4096;  // 16 x 192 u_cat
constexpr int L_MASK = 7168;  // 256 masks
constexpr int L_AO   = 7424;  // 16 x 64 attention out (pre-Wo)
constexpr int L_Z    = 8448;  // 16 x 64 z
constexpr int L_POOL = 9472;  // 6144: KH/VH (f16), then GX|GH, then EK spans AO..end
constexpr int L_TOT  = 15616;

// ---------------- WMMA helpers ------------------------------------------
__device__ __forceinline__ v8f wmma16(v16h a, v16h b, v8f c) {
  return __builtin_amdgcn_wmma_f32_16x16x32_f16(false, a, false, b, (short)0, c, false, false);
}

// A-frag (16x32 f16, M x K) from a zero-padded LDS f32 matrix: per lane this
// is two contiguous 8-float runs -> 4 unconditional ds_load_b128 + pk cvt.
__device__ __forceinline__ v16h afrag(const float* src, int stride, int k0, int lane) {
  int m = lane & 15, grp = lane >> 4;
  const float* rp = src + m * stride + k0 + grp * 8;
  float4 x0 = *(const float4*)(rp + 0);
  float4 x1 = *(const float4*)(rp + 4);
  float4 x2 = *(const float4*)(rp + 16);
  float4 x3 = *(const float4*)(rp + 20);
  v16h a;
  a[0]=(h16)x0.x;  a[1]=(h16)x0.y;  a[2]=(h16)x0.z;  a[3]=(h16)x0.w;
  a[4]=(h16)x1.x;  a[5]=(h16)x1.y;  a[6]=(h16)x1.z;  a[7]=(h16)x1.w;
  a[8]=(h16)x2.x;  a[9]=(h16)x2.y;  a[10]=(h16)x2.z; a[11]=(h16)x2.w;
  a[12]=(h16)x3.x; a[13]=(h16)x3.y; a[14]=(h16)x3.z; a[15]=(h16)x3.w;
  return a;
}

// B-frag (32x16 f16): contiguous 32B per lane, global address space.
__device__ __forceinline__ v16h bfrag(const h16* w, int frag, int lane) {
  return *(gv16p)(w + (size_t)frag * 512 + lane * 16);
}

// GEMM tile with pre-built (hoisted) A fragments; only B streams.
template<int NKF>
__device__ __forceinline__ v8f gemm16a(const v16h* a, const h16* w, int nt, int lane) {
  v8f c = {};
#pragma unroll
  for (int kf = 0; kf < NKF; ++kf)
    c = wmma16(a[kf], bfrag(w, nt * NKF + kf, lane), c);
  return c;
}

// D (16x16 f32) store: VGPR r, lane l -> row r + 8*(l>=16), col l%16
__device__ __forceinline__ void stD(float* dst, int stride, int n0, v8f d, int lane) {
  int n = lane & 15, grp = lane >> 4;
#pragma unroll
  for (int r = 0; r < 8; ++r) dst[(r + grp * 8) * stride + n0 + n] = d[r];
}
__device__ __forceinline__ void stDh(h16* dst, int stride, int row0, int n0, v8f d, int lane) {
  int n = lane & 15, grp = lane >> 4;
#pragma unroll
  for (int r = 0; r < 8; ++r) dst[(row0 + r + grp * 8) * stride + n0 + n] = (h16)d[r];
}

struct Params {
  const float *own, *ally, *enemy, *hidden;
  const float *hyp_b1, *hyp_b2, *ba, *be, *gbih, *gbhh, *bm;
  const h16   *ws;
  float *outQ;   // BN x 22
  float *outZ;   // BN x 64
};

// ---------------- fused MHA (per head loop keeps LDS tiny) ---------------
template<int NR>
__device__ void mha(const float* raw, const h16* wE, const float* bE,
                    const h16* wQ, const h16* wK, const h16* wV, const h16* wO,
                    float* lds, int ucol, int lane) {
  float* SCR  = lds + L_SCR;
  float* OWNE = lds + L_OWNE;
  float* Q    = lds + L_Q;
  float* AO   = lds + L_AO;
  float* U    = lds + L_U;
  float* MASK = lds + L_MASK;
  h16*   KH   = (h16*)(lds + L_POOL);          // 256 x 16
  h16*   VH   = (h16*)(lds + L_POOL + 2048);   // 256 x 16
  const int n = lane & 15;

  // q = own_e @ Wq  (A frags hoisted across the 4 N-tiles)
  {
    v16h aq[2] = { afrag(OWNE, 64, 0, lane), afrag(OWNE, 64, 32, lane) };
#pragma unroll
    for (int nt = 0; nt < 4; ++nt) stD(Q, 64, nt * 16, gemm16a<2>(aq, wQ, nt, lane), lane);
  }
  __syncthreads();

  for (int h = 0; h < NHEAD; ++h) {
    for (int mt = 0; mt < NR; ++mt) {           // NR tiles of 16 (b,n)-rows
      // stage 16 rows x 24 feats, zero cols 24..31 (pad for K=32 frag)
      for (int idx = lane; idx < 16 * 32; idx += 32) {
        int r = idx >> 5, c = idx & 31;
        SCR[r * 64 + c] = (c < FFEAT)
            ? gload(raw + (size_t)(mt * 16 + r) * FFEAT + c) : 0.f;
      }
      __syncthreads();
      if (h == 0 && lane < 16) {
        float m = 0.f;
#pragma unroll
        for (int c = 0; c < FFEAT; ++c) if (SCR[lane * 64 + c] != 0.f) m = 1.f;
        MASK[mt * 16 + lane] = m;
      }
      __syncthreads();
      v8f et[4];
      {
        v16h ae = afrag(SCR, 64, 0, lane);      // one K-frag, 4 N-tiles
#pragma unroll
        for (int nt = 0; nt < 4; ++nt) {
          v8f d = gemm16a<1>(&ae, wE, nt, lane);
          float bb = gload(bE + nt * 16 + n);
#pragma unroll
          for (int r = 0; r < 8; ++r) d[r] += bb;
          et[nt] = d;
        }
      }
      __syncthreads();
#pragma unroll
      for (int nt = 0; nt < 4; ++nt) stD(SCR, 64, nt * 16, et[nt], lane);
      __syncthreads();
      {
        v16h akv[2] = { afrag(SCR, 64, 0, lane), afrag(SCR, 64, 32, lane) };
        stDh(KH, 16, mt * 16, 0, gemm16a<2>(akv, wK, h, lane), lane);
        stDh(VH, 16, mt * 16, 0, gemm16a<2>(akv, wV, h, lane), lane);
      }
      __syncthreads();
    }
    // score / softmax / weighted sum for this head (lanes 0..15 = batch)
    if (lane < 16) {
      int b = lane;
      float q[DHEAD];
#pragma unroll
      for (int j = 0; j < DHEAD; ++j) q[j] = Q[b * 64 + h * 16 + j];
      float s[NR]; float mx = -1e30f;
#pragma unroll
      for (int kk = 0; kk < NR; ++kk) {
        int rr = b * NR + kk;
        float dot = 0.f;
#pragma unroll
        for (int j = 0; j < DHEAD; ++j) dot += q[j] * (float)KH[rr * 16 + j];
        dot *= 0.25f;                        // 1/sqrt(16)
        if (MASK[rr] == 0.f) dot = -1e9f;
        s[kk] = dot; if (dot > mx) mx = dot;
      }
      float den = 0.f;
#pragma unroll
      for (int kk = 0; kk < NR; ++kk) { float e = __expf(s[kk] - mx); s[kk] = e; den += e; }
      float inv = 1.f / den;
#pragma unroll
      for (int j = 0; j < DHEAD; ++j) {
        float o = 0.f;
#pragma unroll
        for (int kk = 0; kk < NR; ++kk) o += s[kk] * (float)VH[(b * NR + kk) * 16 + j];
        AO[b * 64 + h * 16 + j] = o * inv;
      }
    }
    __syncthreads();
  }
  // output projection into u_cat columns [ucol, ucol+64)
  {
    v16h ao[2] = { afrag(AO, 64, 0, lane), afrag(AO, 64, 32, lane) };
#pragma unroll
    for (int nt = 0; nt < 4; ++nt)
      stD(U + ucol, 192, nt * 16, gemm16a<2>(ao, wO, nt, lane), lane);
  }
  __syncthreads();
}

// ---------------- main fused kernel: one wave32 = 16 batch rows ----------
__global__ __launch_bounds__(32) void agent_fused(Params p) {
  __shared__ float lds[L_TOT];
  const int lane = threadIdx.x;
  const int wb0  = blockIdx.x * TB;
  const int n = lane & 15, grp = lane >> 4;

  float* OWN  = lds + L_OWN;
  float* SCR  = lds + L_SCR;
  float* OWNE = lds + L_OWNE;
  float* Q    = lds + L_Q;
  float* U    = lds + L_U;
  float* Z    = lds + L_Z;
  float* GX   = lds + L_POOL;
  float* GH   = lds + L_POOL + 3072;
  h16*   EK   = (h16*)(lds + L_AO);   // 256x64 f16 overlays AO|Z|POOL (free by then)

  // ---- stage own_vec (16 x 40 -> 16 x 64, zero padded) ----
  for (int idx = lane; idx < 16 * 64; idx += 32) {
    int b = idx >> 6, c = idx & 63;
    OWN[idx] = (c < F_OWN) ? gload(p.own + (size_t)(wb0 + b) * F_OWN + c) : 0.f;
  }
  __syncthreads();

  // ---- h1 = relu(own @ hyp_W1 + b1) -> SCR ----
  {
    v16h aw[2] = { afrag(OWN, 64, 0, lane), afrag(OWN, 64, 32, lane) };
    v8f d[4];
#pragma unroll
    for (int nt = 0; nt < 4; ++nt) {
      v8f t = gemm16a<2>(aw, p.ws + OFF_HYPW1, nt, lane);
      float bb = gload(p.hyp_b1 + nt * 16 + n);
#pragma unroll
      for (int r = 0; r < 8; ++r) { float v = t[r] + bb; t[r] = v > 0.f ? v : 0.f; }
      d[nt] = t;
    }
    __syncthreads();
#pragma unroll
    for (int nt = 0; nt < 4; ++nt) stD(SCR, 64, nt * 16, d[nt], lane);
  }
  __syncthreads();

  // ---- hyper GEMM (16x2624 over 164 tiles), contracted on the fly ----
  v8f acc[4]; acc[0] = v8f{}; acc[1] = v8f{}; acc[2] = v8f{}; acc[3] = v8f{};
  {
    v16h a0 = afrag(SCR, 64, 0, lane);
    v16h a1 = afrag(SCR, 64, 32, lane);
    const h16* w2 = p.ws + OFF_HYPW2;
    for (int f = 0; f < F_OWN; ++f) {
      __builtin_prefetch(w2 + (size_t)(f + 2) * 4 * 2 * 512 + lane * 16, 0, 1);
      float coef[8];
#pragma unroll
      for (int r = 0; r < 8; ++r) coef[r] = OWN[(r + grp * 8) * 64 + f];
#pragma unroll
      for (int sub = 0; sub < 4; ++sub) {
        int t = f * 4 + sub;
        v8f d = {};
        d = wmma16(a0, bfrag(w2, t * 2 + 0, lane), d);
        d = wmma16(a1, bfrag(w2, t * 2 + 1, lane), d);
        float bb = gload(p.hyp_b2 + t * 16 + n);   // b2 folds into W_flat
#pragma unroll
        for (int r = 0; r < 8; ++r) acc[sub][r] += coef[r] * (d[r] + bb);
      }
    }
#pragma unroll
    for (int sub = 0; sub < 4; ++sub) {            // b_own tiles (cols 2560..2623)
      int t = 160 + sub;
      v8f d = {};
      d = wmma16(a0, bfrag(w2, t * 2 + 0, lane), d);
      d = wmma16(a1, bfrag(w2, t * 2 + 1, lane), d);
      float bb = gload(p.hyp_b2 + t * 16 + n);
#pragma unroll
      for (int r = 0; r < 8; ++r) acc[sub][r] += d[r] + bb;
    }
  }
  __syncthreads();
#pragma unroll
  for (int sub = 0; sub < 4; ++sub) {
    stD(OWNE, 64, sub * 16, acc[sub], lane);
    stD(U, 192, sub * 16, acc[sub], lane);         // u_cat[:, 0:64] = own_e
  }
  __syncthreads();

  // ---- two attentions ----
  mha<NAL>(p.ally  + (size_t)wb0 * NAL * FFEAT, p.ws + OFF_WA, p.ba,
           p.ws + OFF_AWQ, p.ws + OFF_AWK, p.ws + OFF_AWV, p.ws + OFF_AWO,
           lds, 64, lane);
  mha<NEN>(p.enemy + (size_t)wb0 * NEN * FFEAT, p.ws + OFF_WE, p.be,
           p.ws + OFF_EWQ, p.ws + OFF_EWK, p.ws + OFF_EWV, p.ws + OFF_EWO,
           lds, 128, lane);

  // ---- GRU: stage hidden (16x64 contiguous) via Tensor Data Mover ----
#if __has_builtin(__builtin_amdgcn_tensor_load_to_lds)
  {
    unsigned long long ga = (unsigned long long)(size_t)(p.hidden + (size_t)wb0 * 64);
    u32x4 g0;
    g0[0] = 1u;                                   // count=1, user descriptor
    g0[1] = lds_addr_of(SCR);                     // lds_addr (bytes)
    g0[2] = (unsigned)(ga & 0xffffffffu);         // global_addr[31:0]
    g0[3] = (unsigned)((ga >> 32) & 0x01ffffffu)  // global_addr[56:32]
          | (2u << 30);                           // type = 2 ("image")
    i32x8 g1;
    g1[0] = (int)(2u << 16);    // data_size = 4B, workgroup_mask = 0
    g1[1] = (int)(64u << 16);   // tensor_dim0 = 64 elems
    g1[2] = (int)(16u << 16);   // tensor_dim1 = 16 rows
    g1[3] = (int)(64u << 16);   // tile_dim0 = 64
    g1[4] = 16;                 // tile_dim1 = 16 (tile_dim2 = 0)
    g1[5] = 64;                 // tensor_dim0_stride = 64 elems
    g1[6] = 0; g1[7] = 0;
    i32x4 zz; zz[0] = 0; zz[1] = 0; zz[2] = 0; zz[3] = 0;
#if __clang_major__ >= 23
    i32x8 z8; z8[0]=0; z8[1]=0; z8[2]=0; z8[3]=0; z8[4]=0; z8[5]=0; z8[6]=0; z8[7]=0;
    __builtin_amdgcn_tensor_load_to_lds(g0, g1, zz, zz, z8, 0);
#else
    __builtin_amdgcn_tensor_load_to_lds(g0, g1, zz, zz, 0);
#endif
    __builtin_amdgcn_s_wait_tensorcnt(0);
  }
#else
  for (int idx = lane; idx < 16 * 64; idx += 32)
    SCR[idx] = gload(p.hidden + (size_t)(wb0 + (idx >> 6)) * 64 + (idx & 63));
#endif
  __syncthreads();
  {
    v16h ah[2] = { afrag(SCR, 64, 0, lane), afrag(SCR, 64, 32, lane) };
#pragma unroll
    for (int nt = 0; nt < 12; ++nt) {
      v8f d = gemm16a<2>(ah, p.ws + OFF_GWHH, nt, lane);
      float bb = gload(p.gbhh + nt * 16 + n);
#pragma unroll
      for (int r = 0; r < 8; ++r) d[r] += bb;
      stD(GH, 192, nt * 16, d, lane);
    }
  }
  {
    v16h au[6];
#pragma unroll
    for (int kf = 0; kf < 6; ++kf) au[kf] = afrag(U, 192, kf * 32, lane);
#pragma unroll
    for (int nt = 0; nt < 12; ++nt) {
      v8f d = gemm16a<6>(au, p.ws + OFF_GWIH, nt, lane);
      float bb = gload(p.gbih + nt * 16 + n);
#pragma unroll
      for (int r = 0; r < 8; ++r) d[r] += bb;
      stD(GX, 192, nt * 16, d, lane);
    }
  }
  __syncthreads();
  for (int idx = lane; idx < 16 * 64; idx += 32) {
    int b = idx >> 6, j = idx & 63;
    float xr = GX[b * 192 + j], xz = GX[b * 192 + 64 + j], xn = GX[b * 192 + 128 + j];
    float hr = GH[b * 192 + j], hz = GH[b * 192 + 64 + j], hn = GH[b * 192 + 128 + j];
    float rg = 1.f / (1.f + __expf(-(xr + hr)));
    float zg = 1.f / (1.f + __expf(-(xz + hz)));
    float nc = tanhf(xn + rg * hn);
    float zv = (1.f - zg) * nc + zg * SCR[idx];
    Z[idx] = zv;
    p.outZ[(size_t)(wb0 + b) * 64 + j] = zv;
  }
  __syncthreads();

  // ---- logits_move / zk share hoisted Z fragments ----
  {
    v16h az[2] = { afrag(Z, 64, 0, lane), afrag(Z, 64, 32, lane) };
    v8f d = gemm16a<2>(az, p.ws + OFF_WM, 0, lane);
    if (n < 6) {
      float bb = gload(p.bm + n);
#pragma unroll
      for (int r = 0; r < 8; ++r)
        p.outQ[(size_t)(wb0 + r + grp * 8) * 22 + n] = d[r] + bb;
    }
#pragma unroll
    for (int nt = 0; nt < 4; ++nt)
      stD(Q, 64, nt * 16, gemm16a<2>(az, p.ws + OFF_WZK, nt, lane), lane);
  }
  __syncthreads();

  // ---- Ek = enemy_e @ WEK (recompute enemy_e per tile) ----
  {
    const float* eraw = p.enemy + (size_t)wb0 * NEN * FFEAT;
    for (int mt = 0; mt < NEN; ++mt) {
      for (int idx = lane; idx < 16 * 32; idx += 32) {
        int r = idx >> 5, c = idx & 31;
        SCR[r * 64 + c] = (c < FFEAT)
            ? gload(eraw + (size_t)(mt * 16 + r) * FFEAT + c) : 0.f;
      }
      __syncthreads();
      v8f et[4];
      {
        v16h ae = afrag(SCR, 64, 0, lane);
#pragma unroll
        for (int nt = 0; nt < 4; ++nt) {
          v8f d = gemm16a<1>(&ae, p.ws + OFF_WE, nt, lane);
          float bb = gload(p.be + nt * 16 + n);
#pragma unroll
          for (int r = 0; r < 8; ++r) d[r] += bb;
          et[nt] = d;
        }
      }
      __syncthreads();
#pragma unroll
      for (int nt = 0; nt < 4; ++nt) stD(SCR, 64, nt * 16, et[nt], lane);
      __syncthreads();
      {
        v16h a2[2] = { afrag(SCR, 64, 0, lane), afrag(SCR, 64, 32, lane) };
#pragma unroll
        for (int nt = 0; nt < 4; ++nt)
          stDh(EK, 64, mt * 16, nt * 16, gemm16a<2>(a2, p.ws + OFF_WEKW, nt, lane), lane);
      }
      __syncthreads();
    }
  }
  // ---- logits_shoot[b, m] = zk[b] . Ek[b, m] ----
  for (int pidx = lane; pidx < 256; pidx += 32) {
    int b = pidx >> 4, m = pidx & 15;
    float dot = 0.f;
#pragma unroll
    for (int j = 0; j < 64; ++j) dot += Q[b * 64 + j] * (float)EK[(b * 16 + m) * 64 + j];
    p.outQ[(size_t)(wb0 + b) * 22 + 6 + m] = dot;
  }
}

// ---------------- weight -> f16 B-fragment prep ---------------------------
__global__ void prep_frags(const float* __restrict__ src, h16* __restrict__ dst,
                           int kdim, int ndim, int nkf, int ntiles) {
  int total = ntiles * nkf * 512;
  for (int hidx = blockIdx.x * blockDim.x + threadIdx.x; hidx < total;
       hidx += gridDim.x * blockDim.x) {
    int frag = hidx >> 9;
    int lane = (hidx >> 4) & 31;
    int i    = hidx & 15;
    int nt = frag / nkf, kf = frag % nkf;
    int nn = nt * 16 + (lane & 15);
    int k  = kf * 32 + (i < 8 ? (lane >> 4) * 8 + i : 16 + (lane >> 4) * 8 + (i - 8));   // B layout: K = grp*8 + i (i<8), 16 + grp*8 + i-8 (i>=8)
    float v = (k < kdim && nn < ndim) ? src[(size_t)k * ndim + nn] : 0.f;
    dst[hidx] = (h16)v;
  }
}

extern "C" void kernel_launch(void* const* d_in, const int* in_sizes, int n_in,
                              void* d_out, int out_size, void* d_ws, size_t ws_size,
                              hipStream_t stream) {
  (void)in_sizes; (void)n_in; (void)out_size; (void)ws_size;
  const float* own    = (const float*)d_in[0];
  const float* ally   = (const float*)d_in[1];
  const float* enemy  = (const float*)d_in[2];
  const float* hidden = (const float*)d_in[3];
  const float* hypW1  = (const float*)d_in[4];
  const float* hypb1  = (const float*)d_in[5];
  const float* hypW2  = (const float*)d_in[6];
  const float* hypb2  = (const float*)d_in[7];
  const float* Wa     = (const float*)d_in[8];
  const float* ba     = (const float*)d_in[9];
  const float* We     = (const float*)d_in[10];
  const float* be     = (const float*)d_in[11];
  const float* aWq    = (const float*)d_in[12];
  const float* aWk    = (const float*)d_in[13];
  const float* aWv    = (const float*)d_in[14];
  const float* aWo    = (const float*)d_in[15];
  const float* eWq    = (const float*)d_in[16];
  const float* eWk    = (const float*)d_in[17];
  const float* eWv    = (const float*)d_in[18];
  const float* eWo    = (const float*)d_in[19];
  const float* gWih   = (const float*)d_in[20];
  const float* gWhh   = (const float*)d_in[21];
  const float* gbih   = (const float*)d_in[22];
  const float* gbhh   = (const float*)d_in[23];
  const float* Wm     = (const float*)d_in[24];
  const float* bm     = (const float*)d_in[25];
  const float* WzK    = (const float*)d_in[26];
  const float* WEKw   = (const float*)d_in[27];

  h16* ws = (h16*)d_ws;
  auto prep = [&](const float* s, int off, int kd, int nd, int nkf, int nt) {
    int total = nt * nkf * 512;
    prep_frags<<<(total + 255) / 256, 256, 0, stream>>>(s, ws + off, kd, nd, nkf, nt);
  };
  prep(hypW1, OFF_HYPW1, 40, 64, 2, 4);
  prep(hypW2, OFF_HYPW2, 64, 2628, 2, 164);
  prep(Wa,    OFF_WA,    24, 64, 1, 4);
  prep(We,    OFF_WE,    24, 64, 1, 4);
  prep(aWq,   OFF_AWQ,   64, 64, 2, 4);
  prep(aWk,   OFF_AWK,   64, 64, 2, 4);
  prep(aWv,   OFF_AWV,   64, 64, 2, 4);
  prep(aWo,   OFF_AWO,   64, 64, 2, 4);
  prep(eWq,   OFF_EWQ,   64, 64, 2, 4);
  prep(eWk,   OFF_EWK,   64, 64, 2, 4);
  prep(eWv,   OFF_EWV,   64, 64, 2, 4);
  prep(eWo,   OFF_EWO,   64, 64, 2, 4);
  prep(gWih,  OFF_GWIH, 192, 192, 6, 12);
  prep(gWhh,  OFF_GWHH,  64, 192, 2, 12);
  prep(Wm,    OFF_WM,    64, 6, 2, 1);
  prep(WzK,   OFF_WZK,   64, 64, 2, 4);
  prep(WEKw,  OFF_WEKW,  64, 64, 2, 4);

  Params p;
  p.own = own; p.ally = ally; p.enemy = enemy; p.hidden = hidden;
  p.hyp_b1 = hypb1; p.hyp_b2 = hypb2; p.ba = ba; p.be = be;
  p.gbih = gbih; p.gbhh = gbhh; p.bm = bm;
  p.ws = ws;
  p.outQ = (float*)d_out;
  p.outZ = (float*)d_out + (size_t)BNTOT * 22;

  agent_fused<<<BNTOT / TB, 32, 0, stream>>>(p);
}
